// VideoPrismAttention_33328946217695
// MI455X (gfx1250) — matrix operand order repaired and checked by reading, not compiled, not run on hardware
//
#include <hip/hip_runtime.h>

// ---------------------------------------------------------------------------
// VideoPrism attention for gfx1250 (MI455X): bf16 WMMA everywhere.
// B=4, T=1024, D=1024, N=16 heads, H=64. All GEMMs via v_wmma_f32_16x16x32_bf16.
// ---------------------------------------------------------------------------

typedef __attribute__((ext_vector_type(16))) __bf16 bf16x16;
typedef __attribute__((ext_vector_type(8)))  float  floatx8;

union AFrag { bf16x16 v; unsigned short u[16]; uint4 q[2]; };
union CFrag { floatx8 v; float f[8]; };

__device__ inline unsigned short f32_to_bf16(float f) {
  union { float f; unsigned int u; } c; c.f = f;
  unsigned int u = c.u;
  unsigned int r = u + 0x7FFFu + ((u >> 16) & 1u);   // round-to-nearest-even
  return (unsigned short)(r >> 16);
}

__device__ inline float softcap50(float s) {
  // 50 * tanh(s/50);  tanh(x) = 1 - 2/(e^{2x}+1), rcp instead of IEEE divide
  float x = fminf(fmaxf(s * 0.04f, -30.f), 30.f);    // 2*s/50
  float e = __expf(x);
  return 50.f - 100.f * __builtin_amdgcn_rcpf(e + 1.f);
}

// ---------------------------------------------------------------------------
// f32 -> bf16 elementwise convert (4 elems / thread)
// ---------------------------------------------------------------------------
__global__ void cvt_bf16_kernel(const float* __restrict__ in,
                                unsigned short* __restrict__ out) {
  int i = blockIdx.x * blockDim.x + threadIdx.x;
  float4 v = ((const float4*)in)[i];
  ushort4 o;
  o.x = f32_to_bf16(v.x); o.y = f32_to_bf16(v.y);
  o.z = f32_to_bf16(v.z); o.w = f32_to_bf16(v.w);
  ((ushort4*)out)[i] = o;
}

// ---------------------------------------------------------------------------
// 1024x1024 transpose + f32->bf16:  out[j*1024+i] = bf16(in[i*1024+j])
// ---------------------------------------------------------------------------
__global__ void transpose_cvt_kernel(const float* __restrict__ in,
                                     unsigned short* __restrict__ out) {
  __shared__ float tile[32][33];
  const int bx = blockIdx.x * 32, by = blockIdx.y * 32;
  const int tx = threadIdx.x;
  for (int i = threadIdx.y; i < 32; i += 8)
    tile[i][tx] = in[(by + i) * 1024 + bx + tx];
  __syncthreads();
  for (int i = threadIdx.y; i < 32; i += 8)
    out[(bx + i) * 1024 + by + tx] = f32_to_bf16(tile[tx][i]);
}

// ---------------------------------------------------------------------------
// WMMA GEMM:  C[4096 x 1024] = A[4096 x 1024] * BT^T   (BT is [col][k], bf16)
// Register-blocked: each wave owns a 16x64 output tile (4 accumulators,
// one shared A-fragment per K-step -> 4 independent WMMAs back to back).
//  MODE 0 (Q): out bf16, layout [b,n,t,h], scaled by per-dim softplus scale
//  MODE 1 (K): out bf16, layout [b,n,s,h]
//  MODE 2 (V): out bf16, layout [b,n,h,s]   (transposed for P*V B-frags)
//  MODE 3 (O): out f32,  layout [b,t,d] (+bias)
// ---------------------------------------------------------------------------
template <int MODE>
__global__ void proj_kernel(const unsigned short* __restrict__ A,
                            const unsigned short* __restrict__ BT,
                            const float* __restrict__ bias,
                            const float* __restrict__ pds,
                            void* __restrict__ Out) {
  const int KDIM = 1024;
  const int lane = threadIdx.x & 31;
  const int wave = threadIdx.x >> 5;
  const int tile = blockIdx.x * 8 + wave;        // 256 row-tiles * 16 col-groups
  const int m0 = (tile >> 4) << 4;               // 16-row tile
  const int cg = (tile & 15) << 6;               // 64-col group
  const int hlf = lane >> 4;
  const int l16 = lane & 15;

  CFrag acc[4];
#pragma unroll
  for (int j = 0; j < 4; ++j)
#pragma unroll
    for (int e = 0; e < 8; ++e) acc[j].f[e] = 0.f;

  const unsigned short* arow = A + (m0 + l16) * KDIM;            // A row (m=l16)
  const unsigned short* brow0 = BT + (cg + l16) * KDIM;          // B rows (n=l16)

#pragma unroll 2
  for (int k0 = 0; k0 < KDIM; k0 += 32) {
    AFrag a;
    a.q[0] = *(const uint4*)(arow + k0 + hlf * 8);        // K = hlf*8 + 0..7
    a.q[1] = *(const uint4*)(arow + k0 + 16 + hlf * 8);   // K = 16 + hlf*8 + 0..7
#pragma unroll
    for (int j = 0; j < 4; ++j) {
      AFrag b;
      const unsigned short* brow = brow0 + j * 16 * KDIM;
      b.q[0] = *(const uint4*)(brow + k0 + hlf * 8);      // K = hlf*8 + 0..7
      b.q[1] = *(const uint4*)(brow + k0 + 16 + hlf * 8); // K = 16 + hlf*8 + 0..7
      acc[j].v = __builtin_amdgcn_wmma_f32_16x16x32_bf16(
          false, a.v, false, b.v, (short)0, acc[j].v, false, false);
    }
  }

#pragma unroll
  for (int j = 0; j < 4; ++j) {
    const int c = cg + j * 16 + l16;
    const float bias_v = bias[c];
    float scale = 1.f;
    if (MODE == 0) {  // r_softplus_0 / sqrt(64) * softplus(pds[h])
      float x = pds[c & 63];
      scale = 0.1803368801f * __logf(1.f + __expf(x));
    }
#pragma unroll
    for (int r = 0; r < 8; ++r) {
      const int m = m0 + r + hlf * 8;          // C-layout: row = r + 8*half
      float val = acc[j].f[r] + bias_v;
      if (MODE == 0) val *= scale;
      if (MODE == 3) {
        ((float*)Out)[m * 1024 + c] = val;
      } else {
        const int b = m >> 10, t = m & 1023;
        const int n = c >> 6,  h = c & 63;
        unsigned short* o = (unsigned short*)Out;
        if (MODE == 2)
          o[((b * 16 + n) * 64 + h) * 1024 + t] = f32_to_bf16(val);  // [b,n,h,s]
        else
          o[((b * 16 + n) * 1024 + t) * 64 + h] = f32_to_bf16(val);  // [b,n,t,h]
      }
    }
  }
}

// ---------------------------------------------------------------------------
// Fused flash attention: per-wave 16-row q tile over full T=1024, soft-cap +
// online softmax, P*V accumulated with WMMA. P re-staged via per-wave LDS.
//  qh [b,n,t,h] bf16, kh [b,n,s,h] bf16, vh [b,n,h,s] bf16 -> ctx [b,t,n*h] bf16
// ---------------------------------------------------------------------------
__global__ void attn_kernel(const unsigned short* __restrict__ qh,
                            const unsigned short* __restrict__ kh,
                            const unsigned short* __restrict__ vh,
                            unsigned short* __restrict__ ctx) {
  __shared__ unsigned short Pt[8][16][32];      // per-wave P tile (8 KB)
  const int lane = threadIdx.x & 31;
  const int wave = threadIdx.x >> 5;
  const int gtile = blockIdx.x * 8 + wave;      // 4*16*64 = 4096 tiles
  const int bn = gtile >> 6;                    // (b*16+n)
  const int t0 = (gtile & 63) << 4;
  const int hlf = lane >> 4;
  const int l16 = lane & 15;

  const unsigned short* qbase = qh + (bn * 1024 + t0) * 64;
  const unsigned short* kbase = kh + bn * 1024 * 64;
  const unsigned short* vbase = vh + bn * 64 * 1024;

  // q A-frags, K = h in [0,32) and [32,64)
  AFrag aq0, aq1;
  {
    const unsigned short* qrow = qbase + l16 * 64;
    aq0.q[0] = *(const uint4*)(qrow + hlf * 8);
    aq0.q[1] = *(const uint4*)(qrow + 16 + hlf * 8);
    aq1.q[0] = *(const uint4*)(qrow + 32 + hlf * 8);
    aq1.q[1] = *(const uint4*)(qrow + 48 + hlf * 8);
  }

  CFrag acc[4];
#pragma unroll
  for (int j = 0; j < 4; ++j)
#pragma unroll
    for (int e = 0; e < 8; ++e) acc[j].f[e] = 0.f;
  float rmax[8], rsum[8];
#pragma unroll
  for (int r = 0; r < 8; ++r) { rmax[r] = -1e30f; rsum[r] = 0.f; }

#pragma unroll 1
  for (int s0 = 0; s0 < 1024; s0 += 32) {
    // prefetch next chunk's K rows / V rows into caches (global_prefetch_b8)
    if (s0 + 32 < 1024) {
      __builtin_prefetch(kbase + (s0 + 32 + l16) * 64, 0, 3);
      __builtin_prefetch(kbase + (s0 + 48 + l16) * 64, 0, 3);
      __builtin_prefetch(vbase + l16 * 1024 + s0 + 32, 0, 3);
    }

    // ---- scores: two 16x16 tiles (cols s0..s0+15, s0+16..s0+31), K=H=64 ----
    CFrag sc0, sc1;
#pragma unroll
    for (int e = 0; e < 8; ++e) { sc0.f[e] = 0.f; sc1.f[e] = 0.f; }
    {
      AFrag b;
      const unsigned short* krow0 = kbase + (s0 + l16) * 64;        // row s
      b.q[0] = *(const uint4*)(krow0 + hlf * 8);
      b.q[1] = *(const uint4*)(krow0 + 16 + hlf * 8);
      sc0.v = __builtin_amdgcn_wmma_f32_16x16x32_bf16(
          false, aq0.v, false, b.v, (short)0, sc0.v, false, false);
      b.q[0] = *(const uint4*)(krow0 + 32 + hlf * 8);
      b.q[1] = *(const uint4*)(krow0 + 48 + hlf * 8);
      sc0.v = __builtin_amdgcn_wmma_f32_16x16x32_bf16(
          false, aq1.v, false, b.v, (short)0, sc0.v, false, false);
      const unsigned short* krow1 = krow0 + 16 * 64;
      b.q[0] = *(const uint4*)(krow1 + hlf * 8);
      b.q[1] = *(const uint4*)(krow1 + 16 + hlf * 8);
      sc1.v = __builtin_amdgcn_wmma_f32_16x16x32_bf16(
          false, aq0.v, false, b.v, (short)0, sc1.v, false, false);
      b.q[0] = *(const uint4*)(krow1 + 32 + hlf * 8);
      b.q[1] = *(const uint4*)(krow1 + 48 + hlf * 8);
      sc1.v = __builtin_amdgcn_wmma_f32_16x16x32_bf16(
          false, aq1.v, false, b.v, (short)0, sc1.v, false, false);
    }

    // ---- soft cap + online softmax (row m = r + 8*hlf, col n = l16) ----
#pragma unroll
    for (int r = 0; r < 8; ++r) {
      float v0 = softcap50(sc0.f[r]);
      float v1 = softcap50(sc1.f[r]);
      float mx = fmaxf(v0, v1);
      mx = fmaxf(mx, __shfl_xor(mx, 1, 32));
      mx = fmaxf(mx, __shfl_xor(mx, 2, 32));
      mx = fmaxf(mx, __shfl_xor(mx, 4, 32));
      mx = fmaxf(mx, __shfl_xor(mx, 8, 32));
      float mnew = fmaxf(rmax[r], mx);
      float alpha = __expf(rmax[r] - mnew);
      rmax[r] = mnew;
      float p0 = __expf(v0 - mnew);
      float p1 = __expf(v1 - mnew);
      float ps = p0 + p1;
      ps += __shfl_xor(ps, 1, 32);
      ps += __shfl_xor(ps, 2, 32);
      ps += __shfl_xor(ps, 4, 32);
      ps += __shfl_xor(ps, 8, 32);
      rsum[r] = rsum[r] * alpha + ps;
#pragma unroll
      for (int j = 0; j < 4; ++j) acc[j].f[r] *= alpha;
      Pt[wave][r + hlf * 8][l16]      = f32_to_bf16(p0);
      Pt[wave][r + hlf * 8][l16 + 16] = f32_to_bf16(p1);
    }

    // ---- reload P in A-frag layout (same-wave LDS, no barrier needed) ----
    AFrag ap;
#pragma unroll
    for (int e = 0; e < 8; ++e) ap.u[e]     = Pt[wave][l16][hlf * 8 + e];
#pragma unroll
    for (int e = 0; e < 8; ++e) ap.u[8 + e] = Pt[wave][l16][16 + hlf * 8 + e];

    // ---- context += P(16x32) x V(32x16) for 4 h-chunks ----
#pragma unroll
    for (int j = 0; j < 4; ++j) {
      AFrag b;
      const unsigned short* vrow = vbase + (j * 16 + l16) * 1024 + s0;  // row h
      b.q[0] = *(const uint4*)(vrow + hlf * 8);
      b.q[1] = *(const uint4*)(vrow + 16 + hlf * 8);
      acc[j].v = __builtin_amdgcn_wmma_f32_16x16x32_bf16(
          false, ap.v, false, b.v, (short)0, acc[j].v, false, false);
    }
  }

  // ---- normalize and store context bf16 as [b, t, n*64 + h] ----
  const int bb = bn >> 4, nn = bn & 15;
#pragma unroll
  for (int r = 0; r < 8; ++r) {
    float inv = __builtin_amdgcn_rcpf(rsum[r]);
    int t = t0 + r + hlf * 8;
    unsigned short* crow = ctx + (bb * 1024 + t) * 1024 + nn * 64;
#pragma unroll
    for (int j = 0; j < 4; ++j)
      crow[j * 16 + l16] = f32_to_bf16(acc[j].f[r] * inv);
  }
}

// ---------------------------------------------------------------------------
extern "C" void kernel_launch(void* const* d_in, const int* in_sizes, int n_in,
                              void* d_out, int out_size, void* d_ws, size_t ws_size,
                              hipStream_t stream) {
  const float* query = (const float*)d_in[0];
  const float* value = (const float*)d_in[1];
  const float* Wq    = (const float*)d_in[2];
  const float* bq    = (const float*)d_in[3];
  const float* Wk    = (const float*)d_in[4];
  const float* bk    = (const float*)d_in[5];
  const float* Wv    = (const float*)d_in[6];
  const float* bv    = (const float*)d_in[7];
  const float* Wo    = (const float*)d_in[8];
  const float* bo    = (const float*)d_in[9];
  const float* pds   = (const float*)d_in[10];
  float* out = (float*)d_out;

  char* ws = (char*)d_ws;
  size_t off = 0;
  auto alloc = [&](size_t bytes) -> char* {
    char* p = ws + off;
    off += (bytes + 255) & ~(size_t)255;
    return p;
  };
  const size_t MB8 = (size_t)8 << 20;   // 4*16*1024*64 bf16
  const size_t MB2 = (size_t)2 << 20;   // 1024*1024 bf16
  unsigned short* qh   = (unsigned short*)alloc(MB8);
  unsigned short* kh   = (unsigned short*)alloc(MB8);
  unsigned short* vh   = (unsigned short*)alloc(MB8);
  unsigned short* ctxh = (unsigned short*)alloc(MB8);
  unsigned short* xq   = (unsigned short*)alloc(MB8);
  unsigned short* xv   = (unsigned short*)alloc(MB8);
  unsigned short* WqT  = (unsigned short*)alloc(MB2);
  unsigned short* WkT  = (unsigned short*)alloc(MB2);
  unsigned short* WvT  = (unsigned short*)alloc(MB2);
  unsigned short* WoT  = (unsigned short*)alloc(MB2);

  // stage inputs + weights to bf16 (weights transposed to [col][k])
  cvt_bf16_kernel<<<4096, 256, 0, stream>>>(query, xq);
  cvt_bf16_kernel<<<4096, 256, 0, stream>>>(value, xv);
  dim3 tb(32, 8), tg(32, 32);
  transpose_cvt_kernel<<<tg, tb, 0, stream>>>(Wq, WqT);
  transpose_cvt_kernel<<<tg, tb, 0, stream>>>(Wk, WkT);
  transpose_cvt_kernel<<<tg, tb, 0, stream>>>(Wv, WvT);
  transpose_cvt_kernel<<<tg, tb, 0, stream>>>(Wo, WoT);

  // q/k/v projections (WMMA), q scaled by per-dim softplus scale
  proj_kernel<0><<<512, 256, 0, stream>>>(xq, WqT, bq, pds, qh);
  proj_kernel<1><<<512, 256, 0, stream>>>(xq, WkT, bk, nullptr, kh);
  proj_kernel<2><<<512, 256, 0, stream>>>(xv, WvT, bv, nullptr, vh);

  // fused soft-capped flash attention (WMMA)
  attn_kernel<<<512, 256, 0, stream>>>(qh, kh, vh, ctxh);

  // output projection (WMMA) + bias, f32 out
  proj_kernel<3><<<512, 256, 0, stream>>>(ctxh, WoT, bo, nullptr, out);
}
